// SimpleLatentPipeline_64725157151030
// MI455X (gfx1250) — matrix operand order repaired and checked by reading, not compiled, not run on hardware
//
#include <hip/hip_runtime.h>
#include <hip/hip_bf16.h>
#include <math.h>

// ---------------------------------------------------------------------------
// Types
// ---------------------------------------------------------------------------
typedef _Float16 h16;
typedef __attribute__((ext_vector_type(16))) _Float16 v16h;
typedef __attribute__((ext_vector_type(8)))  _Float16 v8h;
typedef __attribute__((ext_vector_type(8)))  float    v8f;

#define BM 64
#define BN 128
#define BK 64           // two 32-wide WMMA k-steps per LDS stage
#define BPAD 4          // Bs row padding (halves)

// Problem constants (match reference)
#define B_   8192
#define XD_  4096
#define ZD_  1024
#define XE_  64
#define H_   128
#define NSTEPS 10
// 0.5*STEP = 0.005 ; NOISE*sqrt(STEP) = 0.1
#define GRAD_COEF  0.005f
#define NOISE_COEF 0.1f

// ---------------------------------------------------------------------------
// RNG: PCG hash + Box-Muller (deterministic per (idx, seed))
// ---------------------------------------------------------------------------
__device__ __forceinline__ unsigned pcg_hash(unsigned v) {
  unsigned s = v * 747796405u + 2891336453u;
  unsigned w = ((s >> ((s >> 28u) + 4u)) ^ s) * 277803737u;
  return (w >> 22u) ^ w;
}
__device__ __forceinline__ float gauss_one(unsigned idx, unsigned seed) {
  unsigned a = pcg_hash(idx ^ (seed * 0x9E3779B9u));
  unsigned b = pcg_hash(a + 0x632BE5ABu);
  float u1 = (float)(a >> 8) * (1.0f / 16777216.0f) + (1.0f / 33554432.0f); // (0,1)
  float u2 = (float)(b >> 8) * (1.0f / 16777216.0f);
  return sqrtf(-2.0f * __logf(u1)) * __cosf(6.28318530718f * u2);
}

// ---------------------------------------------------------------------------
// Elementwise helpers
// ---------------------------------------------------------------------------
__global__ void cvt_f32_f16(const float* __restrict__ s, h16* __restrict__ d, int n) {
  int i = blockIdx.x * blockDim.x + threadIdx.x;
  if (i < n) d[i] = (h16)s[i];
}
// src is (R,C) row-major f32; dst is (C,R) row-major f16  (i.e. dst = src^T)
__global__ void transpose_cvt(const float* __restrict__ s, h16* __restrict__ d, int R, int C) {
  int i = blockIdx.x * blockDim.x + threadIdx.x;
  if (i < R * C) {
    int r = i / C, c = i - r * C;
    d[(size_t)c * R + r] = (h16)s[i];
  }
}
__global__ void init_z(float* __restrict__ z, h16* __restrict__ zh, int n, unsigned seed) {
  int i = blockIdx.x * blockDim.x + threadIdx.x;
  if (i < n) {
    float g = gauss_one((unsigned)i, seed);
    z[i]  = g;
    zh[i] = (h16)g;
  }
}

// ---------------------------------------------------------------------------
// Fused WMMA GEMM:  C(MxN,f32-acc) = A_f16(MxK) @ B_f16(KxN)  (+bias) +epilogue
//  - 256 threads = 8 waves (4x2), block tile 64x128, wave tile 16x64
//  - BK=64 stage, double-buffered LDS; globals for stage s+1 issued before
//    computing stage s; one barrier per stage.
// Epilogue modes:
//  0: out = acc(+bias) -> outF32 and/or outF16
//  1: outF32 = acc(+bias) (pre-act), outF16 = relu(acc+bias)
//  2: y = acc+bias; r = y - aux1[idx]; outF16 = r; if(outF32) outF32 = y
//  3: dh = (aux1[idx] > 0) ? 2*acc : 0; outF16 = dh
//  4: Langevin: z = aux2[idx] - GRAD_COEF*acc + NOISE_COEF*gauss; aux2 <- z;
//     outF16 = (h16)z
// ---------------------------------------------------------------------------
__global__ __launch_bounds__(256)
void gemm_wmma_f16(const h16* __restrict__ A, const h16* __restrict__ Bm,
                   int M, int N, int K,
                   const float* __restrict__ bias, int mode,
                   float* __restrict__ outF32, h16* __restrict__ outF16,
                   const float* __restrict__ aux1, float* __restrict__ aux2,
                   unsigned rngSeed) {
  __shared__ h16 As[2][BM][BK];          // 2 * 64*64*2  = 16 KB
  __shared__ h16 Bs[2][BN][BK + BPAD];   // 2 * 128*68*2 = 34.8 KB  ([n][k])

  const int tid  = threadIdx.x;
  const int lane = tid & 31;
  const int wave = tid >> 5;
  const int wm   = wave & 3;   // 0..3 -> 16-row slice
  const int wn   = wave >> 2;  // 0..1 -> 64-col slice
  const int m0   = blockIdx.x * BM;
  const int n0   = blockIdx.y * BN;

  // loader geometry
  const int arow = tid >> 2;            // 0..63
  const int akc  = (tid & 3) << 4;      // 0,16,32,48 (halves)
  const h16* aBase = A + (size_t)(m0 + arow) * K + akc;

  v8f c0 = {}, c1 = {}, c2 = {}, c3 = {};
  v8h aR[2], bR[4];

  // ---- stage loaders (global -> regs), (regs -> LDS) ----
  auto loadA = [&](int k0) {
    aR[0] = *(const v8h*)(aBase + k0);
    aR[1] = *(const v8h*)(aBase + k0 + 8);
  };
  auto storeA = [&](int buf) {
    *(v8h*)&As[buf][arow][akc]     = aR[0];
    *(v8h*)&As[buf][arow][akc + 8] = aR[1];
  };
  auto loadB = [&](int k0) {
#pragma unroll
    for (int it = 0; it < 4; ++it) {
      int chunk = tid + it * 256;       // 0..1023
      int kr = chunk >> 4;              // 0..63
      int nc = (chunk & 15) << 3;       // 0..120
      int ng = n0 + nc;
      v8h bv = {};
      if (ng < N) bv = *(const v8h*)(Bm + (size_t)(k0 + kr) * N + ng);
      bR[it] = bv;
    }
  };
  auto storeB = [&](int buf) {
#pragma unroll
    for (int it = 0; it < 4; ++it) {
      int chunk = tid + it * 256;
      int kr = chunk >> 4;
      int nc = (chunk & 15) << 3;
#pragma unroll
      for (int j = 0; j < 8; ++j) Bs[buf][nc + j][kr] = bR[it][j];
    }
  };

  // ---- compute one BK=64 stage from LDS buffer `buf` ----
  auto compute = [&](int buf) {
#pragma unroll
    for (int ks = 0; ks < 2; ++ks) {
      const int ko = ks << 5;
      // A fragment (16x32), CDNA5 wave32 16-bit A layout
      union { v16h v; unsigned u[8]; } af, bf0, bf1, bf2, bf3;
      {
        int m  = (lane & 15) + wm * 16;
        int kh = ko + ((lane >> 4) << 3);
#pragma unroll
        for (int j = 0; j < 8; ++j) {
          int k = ((j >> 2) << 4) + kh + ((j & 3) << 1);
          af.u[j] = *(const unsigned*)&As[buf][m][k];
        }
      }
      // All four B fragments first, then 4 back-to-back WMMAs
      {
        int nb = (lane & 15) + wn * 64;
        int kb = ko + ((lane >> 4) << 3);
#pragma unroll
        for (int j = 0; j < 8; ++j) {
          int k = ((j >> 2) << 4) + kb + ((j & 3) << 1);
          bf0.u[j] = *(const unsigned*)&Bs[buf][nb][k];
          bf1.u[j] = *(const unsigned*)&Bs[buf][nb + 16][k];
          bf2.u[j] = *(const unsigned*)&Bs[buf][nb + 32][k];
          bf3.u[j] = *(const unsigned*)&Bs[buf][nb + 48][k];
        }
      }
      c0 = __builtin_amdgcn_wmma_f32_16x16x32_f16(false, af.v, false, bf0.v, (short)0, c0, false, false);
      c1 = __builtin_amdgcn_wmma_f32_16x16x32_f16(false, af.v, false, bf1.v, (short)0, c1, false, false);
      c2 = __builtin_amdgcn_wmma_f32_16x16x32_f16(false, af.v, false, bf2.v, (short)0, c2, false, false);
      c3 = __builtin_amdgcn_wmma_f32_16x16x32_f16(false, af.v, false, bf3.v, (short)0, c3, false, false);
    }
  };

  // ---- software pipeline: prologue ----
  loadA(0); loadB(0);
  storeA(0); storeB(0);
  __syncthreads();

  const int nstages = K / BK;
  for (int s = 0; s < nstages; ++s) {
    const int buf = s & 1;
    if (s + 1 < nstages) {
      loadA((s + 1) * BK);            // globals in flight during compute
      loadB((s + 1) * BK);
      if (s + 2 < nstages) {          // L2 prefetch two stages ahead
        __builtin_prefetch((const void*)(aBase + (s + 2) * BK), 0, 3);
        int ng = n0 + ((tid & 15) << 3);
        if (ng < N)
          __builtin_prefetch((const void*)(Bm + (size_t)((s + 2) * BK + (tid >> 4)) * N + ng), 0, 3);
      }
    }
    compute(buf);
    if (s + 1 < nstages) {
      storeA(buf ^ 1);
      storeB(buf ^ 1);
    }
    __syncthreads();
  }

  // ---- epilogue: row = wm*16 + (lane/16)*8 + r, col = wn*64 + t*16 + (lane&15)
#pragma unroll
  for (int t = 0; t < 4; ++t) {
    v8f acc = (t == 0) ? c0 : (t == 1) ? c1 : (t == 2) ? c2 : c3;
#pragma unroll
    for (int r = 0; r < 8; ++r) {
      int row  = m0 + wm * 16 + ((lane >> 4) << 3) + r;
      int colg = n0 + wn * 64 + t * 16 + (lane & 15);
      if (colg >= N) continue;
      size_t idx = (size_t)row * N + colg;
      float a = acc[r];
      if (bias) a += bias[colg];
      switch (mode) {
        case 0:
          if (outF32) outF32[idx] = a;
          if (outF16) outF16[idx] = (h16)a;
          break;
        case 1:
          if (outF32) outF32[idx] = a;
          if (outF16) outF16[idx] = (h16)(a > 0.f ? a : 0.f);
          break;
        case 2: {
          float rr = a - aux1[idx];
          outF16[idx] = (h16)rr;
          if (outF32) outF32[idx] = a;
        } break;
        case 3: {
          float dh = (aux1[idx] > 0.f) ? 2.0f * a : 0.0f;
          outF16[idx] = (h16)dh;
        } break;
        case 4: {
          float zo = aux2[idx];
          float g  = gauss_one((unsigned)idx, rngSeed);
          float zn = zo - GRAD_COEF * a + NOISE_COEF * g;
          aux2[idx]  = zn;
          outF16[idx] = (h16)zn;
        } break;
      }
    }
  }
}

// ---------------------------------------------------------------------------
// Fused residual + LayerNorm (H=128), optional final dot-product head.
//   v = a + (bres? bres : 0); o = LN(v)*g + beta
//   if outW: y[row] = dot(o, outW) + outB[0]
// ---------------------------------------------------------------------------
__global__ __launch_bounds__(128)
void ln_fused(const float* __restrict__ a, const float* __restrict__ bres,
              const float* __restrict__ g, const float* __restrict__ beta,
              float* __restrict__ out32, h16* __restrict__ out16,
              const float* __restrict__ outW, const float* __restrict__ outB,
              float* __restrict__ y) {
  __shared__ float red[H_];
  const int row = blockIdx.x;
  const int t   = threadIdx.x;
  const size_t idx = (size_t)row * H_ + t;

  float v = a[idx] + (bres ? bres[idx] : 0.0f);
  red[t] = v; __syncthreads();
  for (int s = H_ / 2; s > 0; s >>= 1) { if (t < s) red[t] += red[t + s]; __syncthreads(); }
  float mean = red[0] * (1.0f / H_); __syncthreads();

  float d = v - mean;
  red[t] = d * d; __syncthreads();
  for (int s = H_ / 2; s > 0; s >>= 1) { if (t < s) red[t] += red[t + s]; __syncthreads(); }
  float var = red[0] * (1.0f / H_); __syncthreads();

  float o = d * rsqrtf(var + 1e-5f) * g[t] + beta[t];
  if (out32) out32[idx] = o;
  if (out16) out16[idx] = (h16)o;

  if (outW) {
    red[t] = o * outW[t]; __syncthreads();
    for (int s = H_ / 2; s > 0; s >>= 1) { if (t < s) red[t] += red[t + s]; __syncthreads(); }
    if (t == 0) y[row] = red[0] + outB[0];
  }
}

// ---------------------------------------------------------------------------
// Host-side launch plumbing
// ---------------------------------------------------------------------------
static inline void launch_gemm(hipStream_t s, const h16* A, const h16* Bm,
                               int M, int N, int K, const float* bias, int mode,
                               float* o32, h16* o16, const float* aux1,
                               float* aux2, unsigned seed) {
  dim3 grid((M + BM - 1) / BM, (N + BN - 1) / BN);
  gemm_wmma_f16<<<grid, 256, 0, s>>>(A, Bm, M, N, K, bias, mode, o32, o16, aux1, aux2, seed);
}
static inline void launch_transpose(hipStream_t s, const float* src, h16* dst, int R, int C) {
  int n = R * C;
  transpose_cvt<<<(n + 255) / 256, 256, 0, s>>>(src, dst, R, C);
}
static inline void launch_cvt(hipStream_t s, const float* src, h16* dst, int n) {
  cvt_f32_f16<<<(n + 255) / 256, 256, 0, s>>>(src, dst, n);
}

extern "C" void kernel_launch(void* const* d_in, const int* in_sizes, int n_in,
                              void* d_out, int out_size, void* d_ws, size_t ws_size,
                              hipStream_t stream) {
  (void)in_sizes; (void)n_in; (void)out_size; (void)ws_size;

  // ---- inputs (all f32), order per setup_inputs()
  const float* x       = (const float*)d_in[0];
  const float* px_W1   = (const float*)d_in[1];
  const float* px_b1   = (const float*)d_in[2];
  const float* px_W2   = (const float*)d_in[3];
  const float* px_b2   = (const float*)d_in[4];
  const float* emb_W   = (const float*)d_in[5];
  const float* emb_b   = (const float*)d_in[6];
  const float* xp_W    = (const float*)d_in[7];
  const float* xp_b    = (const float*)d_in[8];
  const float* zp_W    = (const float*)d_in[9];
  const float* zp_b    = (const float*)d_in[10];
  const float* sa_Wqkv = (const float*)d_in[11];
  const float* sa_bqkv = (const float*)d_in[12];
  const float* sa_Wo   = (const float*)d_in[13];
  const float* sa_bo   = (const float*)d_in[14];
  const float* ca_Wqkv = (const float*)d_in[15];
  const float* ca_bqkv = (const float*)d_in[16];
  const float* ca_Wo   = (const float*)d_in[17];
  const float* ca_bo   = (const float*)d_in[18];
  const float* ffn_W1  = (const float*)d_in[19];
  const float* ffn_b1  = (const float*)d_in[20];
  const float* ffn_W2  = (const float*)d_in[21];
  const float* ffn_b2  = (const float*)d_in[22];
  const float* ln1_g   = (const float*)d_in[23];
  const float* ln1_b   = (const float*)d_in[24];
  const float* ln2_g   = (const float*)d_in[25];
  const float* ln2_b   = (const float*)d_in[26];
  const float* ln3_g   = (const float*)d_in[27];
  const float* ln3_b   = (const float*)d_in[28];
  const float* out_W   = (const float*)d_in[29];
  const float* out_b   = (const float*)d_in[30];

  float* x_recon = (float*)d_out;                       // B*XD
  float* y_pred  = (float*)d_out + (size_t)B_ * XD_;    // B

  // ---- workspace carve-up (~230 MB)
  char* p = (char*)d_ws;
  auto alloc = [&](size_t bytes) -> void* {
    void* r = (void*)p;
    p += (bytes + 255) & ~(size_t)255;
    return r;
  };
  float* zf32  = (float*)alloc((size_t)B_ * ZD_ * 4);
  float* hf32  = (float*)alloc((size_t)B_ * H_ * 4);
  h16*   zf16  = (h16*)  alloc((size_t)B_ * ZD_ * 2);
  h16*   af16  = (h16*)  alloc((size_t)B_ * H_ * 2);
  h16*   rf16  = (h16*)  alloc((size_t)B_ * XD_ * 2);
  h16*   dh16  = (h16*)  alloc((size_t)B_ * H_ * 2);
  h16*   xf16  = (h16*)  alloc((size_t)B_ * XD_ * 2);
  h16*   xe16  = (h16*)  alloc((size_t)B_ * XE_ * 2);
  float* xh32  = (float*)alloc((size_t)B_ * H_ * 4);
  h16*   xh16  = (h16*)  alloc((size_t)B_ * H_ * 2);
  h16*   zh16  = (h16*)  alloc((size_t)B_ * H_ * 2);
  h16*   vsa16 = (h16*)  alloc((size_t)B_ * H_ * 2);
  float* sa32  = (float*)alloc((size_t)B_ * H_ * 4);
  float* h1_32 = (float*)alloc((size_t)B_ * H_ * 4);
  h16*   h1_16 = (h16*)  alloc((size_t)B_ * H_ * 2);
  h16*   vca16 = (h16*)  alloc((size_t)B_ * H_ * 2);
  float* ca32  = (float*)alloc((size_t)B_ * H_ * 4);
  float* h2_32 = (float*)alloc((size_t)B_ * H_ * 4);
  h16*   h2_16 = (h16*)  alloc((size_t)B_ * H_ * 2);
  h16*   ff16  = (h16*)  alloc((size_t)B_ * H_ * 2);
  float* ff32  = (float*)alloc((size_t)B_ * H_ * 4);
  // f16 weights
  h16* W1T  = (h16*)alloc((size_t)ZD_ * H_ * 2);   // (1024x128) = px_W1^T
  h16* W1f  = (h16*)alloc((size_t)H_ * ZD_ * 2);   // (128x1024) = px_W1
  h16* W2T  = (h16*)alloc((size_t)H_ * XD_ * 2);   // (128x4096) = px_W2^T
  h16* W2f  = (h16*)alloc((size_t)XD_ * H_ * 2);   // (4096x128) = px_W2
  h16* embT = (h16*)alloc((size_t)XD_ * XE_ * 2);  // (4096x64)
  h16* xpT  = (h16*)alloc((size_t)XE_ * H_ * 2);   // (64x128)
  h16* zpT  = (h16*)alloc((size_t)ZD_ * H_ * 2);   // (1024x128)
  h16* WvSaT = (h16*)alloc((size_t)H_ * H_ * 2);
  h16* WoSaT = (h16*)alloc((size_t)H_ * H_ * 2);
  h16* WvCaT = (h16*)alloc((size_t)H_ * H_ * 2);
  h16* WoCaT = (h16*)alloc((size_t)H_ * H_ * 2);
  h16* f1T   = (h16*)alloc((size_t)H_ * H_ * 2);
  h16* f2T   = (h16*)alloc((size_t)H_ * H_ * 2);

  // ---- weight prep (every call: deterministic & stateless)
  launch_transpose(stream, px_W1, W1T, H_, ZD_);
  launch_cvt      (stream, px_W1, W1f, H_ * ZD_);
  launch_transpose(stream, px_W2, W2T, XD_, H_);
  launch_cvt      (stream, px_W2, W2f, XD_ * H_);
  launch_transpose(stream, emb_W, embT, XE_, XD_);
  launch_transpose(stream, xp_W,  xpT,  H_, XE_);
  launch_transpose(stream, zp_W,  zpT,  H_, ZD_);
  launch_transpose(stream, sa_Wqkv + 2 * H_ * H_, WvSaT, H_, H_);  // V slice
  launch_transpose(stream, sa_Wo,  WoSaT, H_, H_);
  launch_transpose(stream, ca_Wqkv + 2 * H_ * H_, WvCaT, H_, H_);
  launch_transpose(stream, ca_Wo,  WoCaT, H_, H_);
  launch_transpose(stream, ffn_W1, f1T, H_, H_);
  launch_transpose(stream, ffn_W2, f2T, H_, H_);
  launch_cvt      (stream, x, xf16, B_ * XD_);

  // ---- z0 ~ N(0,1)
  {
    int n = B_ * ZD_;
    init_z<<<(n + 255) / 256, 256, 0, stream>>>(zf32, zf16, n, 0x12345u);
  }

  // ---- Langevin loop: z <- z - 0.5*STEP*grad + NOISE*sqrt(STEP)*g
  for (int s = 0; s < NSTEPS; ++s) {
    // h = z @ W1^T + b1 ; a = relu(h)
    launch_gemm(stream, zf16, W1T, B_, H_, ZD_, px_b1, 1, hf32, af16, nullptr, nullptr, 0);
    // y = a @ W2^T + b2 ; r = y - x
    launch_gemm(stream, af16, W2T, B_, XD_, H_, px_b2, 2, nullptr, rf16, x, nullptr, 0);
    // dh = 2*(r @ W2) * (h > 0)
    launch_gemm(stream, rf16, W2f, B_, H_, XD_, nullptr, 3, nullptr, dh16, hf32, nullptr, 0);
    // dz = dh @ W1 ; z update + noise
    launch_gemm(stream, dh16, W1f, B_, ZD_, H_, nullptr, 4, nullptr, zf16, nullptr, zf32,
                0xBEEF0000u + (unsigned)s);
  }

  // ---- x_recon = px(z_final)
  launch_gemm(stream, zf16, W1T, B_, H_, ZD_, px_b1, 1, hf32, af16, nullptr, nullptr, 0);
  launch_gemm(stream, af16, W2T, B_, XD_, H_, px_b2, 2, x_recon, rf16, x, nullptr, 0);

  // ---- transformer head (seq len 1 -> attention == V/O projections)
  // x_e = x @ emb_W^T + emb_b
  launch_gemm(stream, xf16, embT, B_, XE_, XD_, emb_b, 0, nullptr, xe16, nullptr, nullptr, 0);
  // x_h = x_e @ xp_W^T + xp_b
  launch_gemm(stream, xe16, xpT, B_, H_, XE_, xp_b, 0, xh32, xh16, nullptr, nullptr, 0);
  // z_h = z @ zp_W^T + zp_b
  launch_gemm(stream, zf16, zpT, B_, H_, ZD_, zp_b, 0, nullptr, zh16, nullptr, nullptr, 0);
  // sa = (x_h @ Wv^T + bv) @ Wo^T + bo
  launch_gemm(stream, xh16, WvSaT, B_, H_, H_, sa_bqkv + 2 * H_, 0, nullptr, vsa16, nullptr, nullptr, 0);
  launch_gemm(stream, vsa16, WoSaT, B_, H_, H_, sa_bo, 0, sa32, nullptr, nullptr, nullptr, 0);
  // h1 = LN(x_h + sa)
  ln_fused<<<B_, H_, 0, stream>>>(xh32, sa32, ln1_g, ln1_b, h1_32, h1_16,
                                  nullptr, nullptr, nullptr);
  // ca = (z_h @ Wv^T + bv) @ Wo^T + bo
  launch_gemm(stream, zh16, WvCaT, B_, H_, H_, ca_bqkv + 2 * H_, 0, nullptr, vca16, nullptr, nullptr, 0);
  launch_gemm(stream, vca16, WoCaT, B_, H_, H_, ca_bo, 0, ca32, nullptr, nullptr, nullptr, 0);
  // h2 = LN(h1 + ca)
  ln_fused<<<B_, H_, 0, stream>>>(h1_32, ca32, ln2_g, ln2_b, h2_32, h2_16,
                                  nullptr, nullptr, nullptr);
  // ff = relu(h2 @ W1^T + b1) @ W2^T + b2
  launch_gemm(stream, h2_16, f1T, B_, H_, H_, ffn_b1, 1, nullptr, ff16, nullptr, nullptr, 0);
  launch_gemm(stream, ff16, f2T, B_, H_, H_, ffn_b2, 0, ff32, nullptr, nullptr, nullptr, 0);
  // h3 = LN(h2 + ff) ; y = h3 @ out_W^T + out_b   (fused)
  ln_fused<<<B_, H_, 0, stream>>>(h2_32, ff32, ln3_g, ln3_b, nullptr, nullptr,
                                  out_W, out_b, y_pred);
}
